// GraphLayer_2001454760696
// MI455X (gfx1250) — matrix operand order repaired and checked by reading, not compiled, not run on hardware
//
#include <hip/hip_runtime.h>
#include <hip/hip_bf16.h>

// B=4, N=256, H=128
#define BN_ 1024          // B*N rows for x-side
#define NN  256
#define HH  128

typedef __attribute__((ext_vector_type(2)))  float  v2f;
typedef __attribute__((ext_vector_type(8)))  float  v8f;
typedef __attribute__((ext_vector_type(16))) __bf16 v16bf;
typedef __attribute__((ext_vector_type(8)))  __bf16 v8bf;

// ---------------------------------------------------------------------------
// CDNA5 async global->LDS copy (ASYNCcnt-tracked) + wait.
// ---------------------------------------------------------------------------
__device__ __forceinline__ void async_copy_b128(unsigned lds_byte_off,
                                                const float* __restrict__ gsrc)
{
    asm volatile("global_load_async_to_lds_b128 %0, %1, off"
                 :: "v"(lds_byte_off),
                    "v"((unsigned long long)(size_t)gsrc)
                 : "memory");
}

__device__ __forceinline__ void wait_async0()
{
#if __has_builtin(__builtin_amdgcn_s_wait_asynccnt)
    __builtin_amdgcn_s_wait_asynccnt(0);
#else
    asm volatile("s_wait_asynccnt 0x0" ::: "memory");
#endif
}

// ---------------------------------------------------------------------------
// K0: zero the BN-e atomic accumulators (sum + sumsq, 256 floats)
// ---------------------------------------------------------------------------
__global__ void k0_zero(float* p) { p[threadIdx.x] = 0.0f; }

// ---------------------------------------------------------------------------
// K1: x-side linears: out = x @ W^T + b for (Ve, Vn, Un) via f32 WMMA 16x16x4.
// grid = (64 row-tiles, 3 matrices), block = 256 (8 waves, wave w -> h-tile w)
// ---------------------------------------------------------------------------
__global__ void __launch_bounds__(256)
k1_xlin(const float* __restrict__ x,
        const float* __restrict__ w0, const float* __restrict__ b0,
        const float* __restrict__ w1, const float* __restrict__ b1,
        const float* __restrict__ w2, const float* __restrict__ b2,
        float* __restrict__ o0, float* __restrict__ o1, float* __restrict__ o2)
{
    const int tile = blockIdx.x;                  // 16-row tile of BN_=1024
    const int mat  = blockIdx.y;                  // 0..2
    const float* __restrict__ w   = (mat == 0) ? w0 : (mat == 1) ? w1 : w2;
    const float* __restrict__ bia = (mat == 0) ? b0 : (mat == 1) ? b1 : b2;
    float* __restrict__ out       = (mat == 0) ? o0 : (mat == 1) ? o1 : o2;

    const int lane = threadIdx.x & 31;
    const int wave = threadIdx.x >> 5;
    const int l15  = lane & 15;
    const int half = lane >> 4;
    const int n    = wave * 16 + l15;             // output channel (B/C n-index)
    const int mrow = tile * 16 + l15;             // A row

    v8f c = {};
    for (int k0 = 0; k0 < HH; k0 += 4) {
        const int ka = k0 + half * 2;
        v2f a, bf;
        a.x  = x[mrow * HH + ka];
        a.y  = x[mrow * HH + ka + 1];
        bf.x = w[n * HH + ka];                    // B[k,n] = W[n,k]
        bf.y = w[n * HH + ka + 1];
        c = __builtin_amdgcn_wmma_f32_16x16x4_f32(false, a, false, bf,
                                                  (short)0, c, false, false);
    }
    const float bv = bia[n];
#pragma unroll
    for (int r = 0; r < 8; ++r) {
        const int m = tile * 16 + r + half * 8;   // C layout: M = r (+8 for hi half)
        out[m * HH + n] = c[r] + bv;
    }
}

// ---------------------------------------------------------------------------
// K2: main fused kernel. One block per (b,i). 8 waves x 16-h tile.
// e_feat = e@Ue^T + Ue_b + Vx_i + Vx_j via bf16 WMMA; fuse sigmoid, the
// einsum (num), gate-sum (den), BN-e partial stats; e_feat -> d_out scratch.
// Vx/Vnx j-tiles arrive via double-buffered async global->LDS copies that
// overlap the previous iteration's WMMA + post-process; the next e tile's
// float4 registers are prefetched across the compute section.
// ---------------------------------------------------------------------------
__global__ void __launch_bounds__(256)
k2_edge(const float* __restrict__ e, const float* __restrict__ Uew,
        const float* __restrict__ Ueb, const float* __restrict__ Vx,
        const float* __restrict__ Vnx, float* __restrict__ efeat,
        float* __restrict__ num, float* __restrict__ den,
        float* __restrict__ bn_sum, float* __restrict__ bn_sumsq)
{
    __shared__ __bf16 atile[16 * HH];             // 4 KB bf16 staging of e tile
    __shared__ float  sVx [2][16 * HH];           // 2 x 8 KB, async-filled
    __shared__ float  sVnx[2][16 * HH];           // 2 x 8 KB, async-filled

    const int bi   = blockIdx.x;                  // b*256 + i
    const int b    = bi >> 8;
    const int tid  = threadIdx.x;
    const int lane = tid & 31;
    const int wave = tid >> 5;
    const int l15  = lane & 15;
    const int half = lane >> 4;
    const int h    = wave * 16 + l15;             // this lane's output channel

    // --- B fragments: Ue^T slice for this h-tile, bf16, kept in VGPRs -------
    v16bf bfrag[4];
#pragma unroll
    for (int kk = 0; kk < 4; ++kk) {
#pragma unroll
        for (int p = 0; p < 8; ++p) {
            const int k = kk * 32 + half * 16 + p * 2;   // lanes 0-15: K lo16, 16-31: K hi16
            bfrag[kk][2 * p]     = (__bf16)Uew[h * HH + k];
            bfrag[kk][2 * p + 1] = (__bf16)Uew[h * HH + k + 1];
        }
    }

    const float binv = Ueb[h] + Vx[bi * HH + h];  // bias + row term, loop invariant
    const float* __restrict__ erow = e     + (size_t)bi * NN * HH;
    float*       __restrict__ frow = efeat + (size_t)bi * NN * HH;
    const float* __restrict__ vxb  = Vx  + (size_t)b * NN * HH;
    const float* __restrict__ vnxb = Vnx + (size_t)b * NN * HH;

    const int srow = (tid * 8) >> 7;              // staging coords: 2048 elems / 256 thr
    const int scol = (tid * 8) & 127;

    // per-thread async-copy slots: two b128 per array (8 KB / 256 thr / 16 B = 2)
    const unsigned vx_lds0  = (unsigned)(size_t)&sVx [0][tid * 4];
    const unsigned vx_lds1  = (unsigned)(size_t)&sVx [1][tid * 4];
    const unsigned vnx_lds0 = (unsigned)(size_t)&sVnx[0][tid * 4];
    const unsigned vnx_lds1 = (unsigned)(size_t)&sVnx[1][tid * 4];

    float numa = 0.f, dena = 0.f, se = 0.f, ss = 0.f;

    // --- prologue: prefetch e tile 0 regs, issue async Vx/Vnx tile 0 --------
    float4 f0, f1;
    {
        const float4* gp = reinterpret_cast<const float4*>(erow + srow * HH + scol);
        f0 = gp[0]; f1 = gp[1];
    }
    async_copy_b128(vx_lds0,          vxb  + tid * 4);
    async_copy_b128(vx_lds0  + 4096,  vxb  + 1024 + tid * 4);
    async_copy_b128(vnx_lds0,         vnxb + tid * 4);
    async_copy_b128(vnx_lds0 + 4096,  vnxb + 1024 + tid * 4);

    for (int t = 0; t < 16; ++t) {
        const int j0  = t * 16;
        const int cur = t & 1;

        wait_async0();                            // this wave's async tile t done
        __syncthreads();                          // all LDS writes visible; prev reads done

        { // stage 16x128 fp32 -> bf16 into LDS (prefetched regs in, b128 out)
            union { v8bf v; uint4 u; } pk;
            pk.v[0] = (__bf16)f0.x; pk.v[1] = (__bf16)f0.y;
            pk.v[2] = (__bf16)f0.z; pk.v[3] = (__bf16)f0.w;
            pk.v[4] = (__bf16)f1.x; pk.v[5] = (__bf16)f1.y;
            pk.v[6] = (__bf16)f1.z; pk.v[7] = (__bf16)f1.w;
            *reinterpret_cast<uint4*>(atile + srow * HH + scol) = pk.u;
        }

        if (t < 15) {                             // async Vx/Vnx tile t+1 -> other buffer
            const float* vxs  = vxb  + (j0 + 16) * HH;
            const float* vnxs = vnxb + (j0 + 16) * HH;
            const unsigned dvx  = cur ? vx_lds0  : vx_lds1;
            const unsigned dvnx = cur ? vnx_lds0 : vnx_lds1;
            async_copy_b128(dvx,          vxs  + tid * 4);
            async_copy_b128(dvx  + 4096,  vxs  + 1024 + tid * 4);
            async_copy_b128(dvnx,         vnxs + tid * 4);
            async_copy_b128(dvnx + 4096,  vnxs + 1024 + tid * 4);
        }

        __syncthreads();                          // e tile ready for A-fragment reads

        if (t < 15) {                             // prefetch next e tile into regs
            const float4* gp = reinterpret_cast<const float4*>(
                erow + (j0 + 16 + srow) * HH + scol);
            f0 = gp[0]; f1 = gp[1];
        }

        v8f c = {};
#pragma unroll
        for (int kk = 0; kk < 4; ++kk) {
            union { v16bf v; uint4 u[2]; } af;
            const uint4* lp = reinterpret_cast<const uint4*>(
                atile + l15 * HH + kk * 32 + half * 16);
            af.u[0] = lp[0];                      // K chunk [half*16 .. +7]
            af.u[1] = lp[1];                      // K chunk [half*16+8 .. +7]
            c = __builtin_amdgcn_wmma_f32_16x16x32_bf16(false, af.v, false, bfrag[kk],
                                                        (short)0, c, false, false);
        }

#pragma unroll
        for (int r = 0; r < 8; ++r) {
            const int jl = r + half * 8;          // C layout: M row = local j
            const int j  = j0 + jl;
            const float f = c[r] + binv + sVx[cur][jl * HH + h];
            const float g = 1.0f / (1.0f + __expf(-f));
            numa += g * sVnx[cur][jl * HH + h];
            dena += g;
            se += f;
            ss += f * f;
            frow[j * HH + h] = f;                 // e_feat scratch (d_out e-region)
        }
    }

    // lanes L and L+16 hold same h, disjoint j-sets -> pairwise reduce
    numa += __shfl_xor(numa, 16);
    dena += __shfl_xor(dena, 16);
    se   += __shfl_xor(se, 16);
    ss   += __shfl_xor(ss, 16);
    if (half == 0) {
        num[bi * HH + h] = numa;
        den[bi * HH + h] = dena;
        atomicAdd(&bn_sum[h], se);
        atomicAdd(&bn_sumsq[h], ss);
    }
}

// ---------------------------------------------------------------------------
// K3: finalize BN-e mean/rstd; full x-path: x_feat = Unx + num/den, BN over
// 1024 rows per channel, relu + residual -> x_new. One block, thread = channel.
// ---------------------------------------------------------------------------
__global__ void k3_xpath(const float* __restrict__ x, const float* __restrict__ Unx,
                         const float* __restrict__ num, const float* __restrict__ den,
                         const float* __restrict__ gx, const float* __restrict__ bx,
                         const float* __restrict__ bn_sum, const float* __restrict__ bn_sumsq,
                         float* __restrict__ mean_e, float* __restrict__ rstd_e,
                         float* __restrict__ xout)
{
    const int h = threadIdx.x;                    // 0..127
    const float cnt_e = (float)(4 * NN * NN);
    const float me = bn_sum[h] / cnt_e;
    const float ve = bn_sumsq[h] / cnt_e - me * me;
    mean_e[h] = me;
    rstd_e[h] = rsqrtf(ve + 1e-5f);

    float s = 0.f, s2 = 0.f;
    for (int r = 0; r < BN_; ++r) {
        const float xf = Unx[r * HH + h] + num[r * HH + h] / (1e-20f + den[r * HH + h]);
        s += xf; s2 += xf * xf;
    }
    const float m  = s / (float)BN_;
    const float v  = s2 / (float)BN_ - m * m;
    const float rs = rsqrtf(v + 1e-5f);
    const float g = gx[h], bt = bx[h];
    for (int r = 0; r < BN_; ++r) {
        const float xf  = Unx[r * HH + h] + num[r * HH + h] / (1e-20f + den[r * HH + h]);
        const float val = (xf - m) * rs * g + bt;
        xout[r * HH + h] = x[r * HH + h] + fmaxf(val, 0.0f);
    }
}

// ---------------------------------------------------------------------------
// K4: e_new = e + relu(bn(e_feat)), float4 in-place over d_out e-region.
// ---------------------------------------------------------------------------
__global__ void __launch_bounds__(256)
k4_eapply(const float* __restrict__ e, const float* __restrict__ mean_e,
          const float* __restrict__ rstd_e, const float* __restrict__ ge,
          const float* __restrict__ be, float* __restrict__ eio)
{
    __shared__ float sm[HH], sr[HH], sg[HH], sb[HH];
    const int t = threadIdx.x;
    if (t < HH) { sm[t] = mean_e[t]; sr[t] = rstd_e[t]; sg[t] = ge[t]; sb[t] = be[t]; }
    __syncthreads();

    const size_t idx = ((size_t)blockIdx.x * 256 + t) * 4;
    const int h = (int)(idx & 127);
    const float4 f  = *reinterpret_cast<const float4*>(eio + idx);
    const float4 ev = *reinterpret_cast<const float4*>(e + idx);
    float4 o;
    o.x = ev.x + fmaxf((f.x - sm[h    ]) * sr[h    ] * sg[h    ] + sb[h    ], 0.0f);
    o.y = ev.y + fmaxf((f.y - sm[h + 1]) * sr[h + 1] * sg[h + 1] + sb[h + 1], 0.0f);
    o.z = ev.z + fmaxf((f.z - sm[h + 2]) * sr[h + 2] * sg[h + 2] + sb[h + 2], 0.0f);
    o.w = ev.w + fmaxf((f.w - sm[h + 3]) * sr[h + 3] * sg[h + 3] + sb[h + 3], 0.0f);
    *reinterpret_cast<float4*>(eio + idx) = o;
}

// ---------------------------------------------------------------------------
extern "C" void kernel_launch(void* const* d_in, const int* in_sizes, int n_in,
                              void* d_out, int out_size, void* d_ws, size_t ws_size,
                              hipStream_t stream)
{
    const float* x   = (const float*)d_in[0];
    const float* e   = (const float*)d_in[1];
    const float* Uew = (const float*)d_in[2];
    const float* Ueb = (const float*)d_in[3];
    const float* Vew = (const float*)d_in[4];
    const float* Veb = (const float*)d_in[5];
    const float* Unw = (const float*)d_in[6];
    const float* Unb = (const float*)d_in[7];
    const float* Vnw = (const float*)d_in[8];
    const float* Vnb = (const float*)d_in[9];
    const float* gx  = (const float*)d_in[10];
    const float* bx  = (const float*)d_in[11];
    const float* ge  = (const float*)d_in[12];
    const float* be  = (const float*)d_in[13];

    float* out   = (float*)d_out;
    float* xout  = out;                 // (B,N,H)
    float* efeat = out + BN_ * HH;      // e-region: e_feat scratch, then e_new in place

    float* ws       = (float*)d_ws;
    float* Vxb      = ws;               // 1024*128
    float* Vnxb     = ws + 131072;
    float* Unxb     = ws + 262144;
    float* numb     = ws + 393216;
    float* denb     = ws + 524288;
    float* bn_sum   = ws + 655360;      // 128
    float* bn_sumsq = bn_sum + 128;     // 128
    float* mean_e   = bn_sum + 256;     // 128
    float* rstd_e   = bn_sum + 384;     // 128

    k0_zero<<<1, 256, 0, stream>>>(bn_sum);

    dim3 g1(BN_ / 16, 3);
    k1_xlin<<<g1, 256, 0, stream>>>(x, Vew, Veb, Vnw, Vnb, Unw, Unb,
                                    Vxb, Vnxb, Unxb);

    k2_edge<<<BN_, 256, 0, stream>>>(e, Uew, Ueb, Vxb, Vnxb, efeat,
                                     numb, denb, bn_sum, bn_sumsq);

    k3_xpath<<<1, 128, 0, stream>>>(x, Unxb, numb, denb, gx, bx,
                                    bn_sum, bn_sumsq, mean_e, rstd_e, xout);

    const int n4blocks = (4 * NN * NN * HH) / (256 * 4);   // 32768
    k4_eapply<<<n4blocks, 256, 0, stream>>>(e, mean_e, rstd_e, ge, be, efeat);
}
